// MemoryCenters_33483565040118
// MI455X (gfx1250) — matrix operand order repaired and checked by reading, not compiled, not run on hardware
//
#include <hip/hip_runtime.h>
#include <hip/hip_bf16.h>
#include <stdint.h>

// ---------------------------------------------------------------------------
// Types
// ---------------------------------------------------------------------------
typedef __bf16 bf16_t;
typedef __bf16 bf16x16 __attribute__((ext_vector_type(16)));
typedef float  f32x8   __attribute__((ext_vector_type(8)));
typedef unsigned int u32x4 __attribute__((ext_vector_type(4)));
typedef int    i32x4   __attribute__((ext_vector_type(4)));
typedef int    i32x8   __attribute__((ext_vector_type(8)));

union FragBF { bf16x16 v; u32x4 u[2]; };

#define TOPK 32
#define D    128
#define QT   16            // queries per tile (WMMA M)
#define ROWS_PER_ITER 64   // K-matrix rows staged per iteration (4 waves x 16)
#define LDS_ROW_BYTES 272  // 256B row + 16B pad (TDM pad: 4 dwords per 64 dwords)
#define SCORE_STRIDE 68
#define NCHUNKS 32

#if __has_builtin(__builtin_amdgcn_tensor_load_to_lds)
#define HAVE_TDM 1
#endif

// ---------------------------------------------------------------------------
// Kernel 0: fp32 -> bf16 conversion (q and K). K_bf16 = 25.6MB -> fits L2.
// ---------------------------------------------------------------------------
__global__ void cvt_bf16_kernel(const float* __restrict__ s,
                                bf16_t* __restrict__ d, int n) {
  int i = blockIdx.x * blockDim.x + threadIdx.x;
  if (i < n) d[i] = (bf16_t)s[i];
}

// ---------------------------------------------------------------------------
// TDM issue: 2D tile, 64 rows x 128 bf16 elems, padded into LDS (stride 272B)
// D# layout per CDNA5 ISA.
// This toolchain's builtin is the 6-arg form:
//   (uint32x4 g0, int32x8 g1, int32x4 g2, int32x4 g3, int32x8 g4, i32 cpol)
// ---------------------------------------------------------------------------
__device__ __forceinline__ void tdm_issue(const bf16_t* gsrc, unsigned ldsOff,
                                          unsigned rowsRemaining) {
#ifdef HAVE_TDM
  unsigned long long ga = (unsigned long long)(uintptr_t)gsrc;
  u32x4 g0;
  g0[0] = 1u;                                   // count=1, user descriptor
  g0[1] = ldsOff;                               // lds_addr (bytes)
  g0[2] = (unsigned)ga;                         // global_addr[31:0]
  g0[3] = (unsigned)((ga >> 32) & 0x01FFFFFFull) | (2u << 30);  // addr hi | type=2
  i32x8 g1;
  // wg_mask=0 | data_size(2B)=1<<16 | pad_enable<<20 | pad_interval(64dw)=5<<22
  // | pad_amount(4dw)=3<<25
  g1[0] = (int)((1u << 16) | (1u << 20) | (5u << 22) | (3u << 25));
  g1[1] = (int)(128u << 16);                    // tensor_dim0 = 128 (bits 79:48)
  g1[2] = (int)((rowsRemaining & 0xFFFFu) << 16);          // tensor_dim1 lo
  g1[3] = (int)((rowsRemaining >> 16) | (128u << 16));     // dim1 hi | tile_dim0=128
  g1[4] = ROWS_PER_ITER;                        // tile_dim1=64, tile_dim2=0
  g1[5] = 128;                                  // tensor_dim0_stride = 128 elems
  g1[6] = 0;
  g1[7] = 0;
  i32x4 z4 = {0, 0, 0, 0};
  i32x8 z8 = {0, 0, 0, 0, 0, 0, 0, 0};
  __builtin_amdgcn_tensor_load_to_lds(g0, g1, z4, z4, z8, 0);
#else
  (void)gsrc; (void)ldsOff; (void)rowsRemaining;
#endif
}

// ---------------------------------------------------------------------------
// Kernel 1: bf16 WMMA similarity + streaming per-query top-32 over a chunk.
// Block: 128 threads (4 waves). Grid: (BT/16, NCHUNKS).
// ---------------------------------------------------------------------------
__global__ __launch_bounds__(128) void sim_topk_kernel(
    const bf16_t* __restrict__ qb, const bf16_t* __restrict__ Kb,
    int N, int chunkSize, int nChunks,
    float* __restrict__ candVal, int* __restrict__ candIdx) {
  __shared__ __align__(16) char qs[QT * LDS_ROW_BYTES];
  __shared__ __align__(16) char Ks[2][ROWS_PER_ITER * LDS_ROW_BYTES];
  __shared__ float scores[QT * SCORE_STRIDE];
  __shared__ float tval[QT][TOPK];
  __shared__ int   tidxs[QT][TOPK];

  const int tid  = threadIdx.x;
  const int wave = tid >> 5;
  const int lrow = tid & 15;        // row-within-tile for frag loads
  const int hi   = (tid >> 4) & 1;  // lane-half selector
  const int qtile = blockIdx.x;
  const int chunk = blockIdx.y;
  const int chunkStart = chunk * chunkSize;
  const int chunkEnd   = min(chunkStart + chunkSize, N);
  const int iters = (chunkEnd - chunkStart + ROWS_PER_ITER - 1) / ROWS_PER_ITER;

  // Stage query tile (16 rows x 256B) into padded LDS rows.
  {
    int r  = tid >> 3;   // 0..15
    int cc = tid & 7;    // two 16B chunks per thread
    const u32x4* src = (const u32x4*)(qb + (size_t)(qtile * QT + r) * D);
    u32x4* dst = (u32x4*)(qs + r * LDS_ROW_BYTES);
    dst[cc]     = src[cc];
    dst[cc + 8] = src[cc + 8];
  }
  if (tid < QT) {
    for (int j = 0; j < TOPK; ++j) { tval[tid][j] = -1e30f; tidxs[tid][j] = 0; }
  }
  __syncthreads();

  // Hoist A fragments: 16x32 bf16 per K-step, reused across all iterations.
  // Layout (ISA 7.12.2): lanes 0-15 K={0..7,16..23}, lanes 16-31 K={8..15,24..31}.
  FragBF afr[4];
  {
    const char* arow = qs + lrow * LDS_ROW_BYTES;
#pragma unroll
    for (int t = 0; t < 4; ++t) {
      afr[t].u[0] = *(const u32x4*)(arow + 64 * t + 16 * hi);
      afr[t].u[1] = *(const u32x4*)(arow + 64 * t + 32 + 16 * hi);
    }
  }

#ifdef HAVE_TDM
  if (tid < 32) {  // wave 0 owns the TDM pipe (EXEC ignored by tensor ops)
    tdm_issue(Kb + (size_t)chunkStart * D,
              (unsigned)(uintptr_t)&Ks[0][0],
              (unsigned)(N - chunkStart));
  }
#endif

  for (int it = 0; it < iters; ++it) {
    const int rowBase = chunkStart + it * ROWS_PER_ITER;
#ifdef HAVE_TDM
    if (tid < 32) {
      if (it + 1 < iters) {
        int nb = rowBase + ROWS_PER_ITER;
        tdm_issue(Kb + (size_t)nb * D,
                  (unsigned)(uintptr_t)&Ks[(it + 1) & 1][0],
                  (unsigned)(N - nb));
        __builtin_amdgcn_s_wait_tensorcnt(1);  // tile `it` complete
      } else {
        __builtin_amdgcn_s_wait_tensorcnt(0);
      }
    }
    __syncthreads();
    const char* kbuf = &Ks[it & 1][0];
#else
    __syncthreads();  // previous readers done before overwrite
    {
      int rows = min(ROWS_PER_ITER, N - rowBase);
      for (int i = tid; i < ROWS_PER_ITER * 16; i += 128) {
        int r = i >> 4, c = i & 15;
        u32x4 val = {0, 0, 0, 0};
        if (r < rows)
          val = *(const u32x4*)((const char*)(Kb + (size_t)(rowBase + r) * D) + c * 16);
        *(u32x4*)(&Ks[0][0] + r * LDS_ROW_BYTES + c * 16) = val;
      }
    }
    __syncthreads();
    const char* kbuf = &Ks[0][0];
#endif

    // 16x16 score tile per wave; K-loop of 4 bf16 WMMAs (K=32 each).
    // B = K-rows (N-major == column-major B): lanes 0-15 K={0..7,16..23}, lanes 16-31 K={8..15,24..31}.
    f32x8 acc = {0.f, 0.f, 0.f, 0.f, 0.f, 0.f, 0.f, 0.f};
    const char* krow = kbuf + (wave * 16 + lrow) * LDS_ROW_BYTES;
#pragma unroll
    for (int t = 0; t < 4; ++t) {
      FragBF b;
      b.u[0] = *(const u32x4*)(krow + 64 * t + 16 * hi);
      b.u[1] = *(const u32x4*)(krow + 64 * t + 32 + 16 * hi);
      acc = __builtin_amdgcn_wmma_f32_16x16x32_bf16(false, afr[t].v, false, b.v,
                                                    (short)0, acc, false, false);
    }
#pragma unroll
    for (int j = 0; j < 8; ++j)
      scores[(j + 8 * hi) * SCORE_STRIDE + wave * 16 + lrow] = acc[j];
    __syncthreads();

    // 16 owner threads: sorted-descending top-32 insertion (cheap after warm-up).
    if (tid < QT) {
      float* tv = tval[tid];
      int*   ti = tidxs[tid];
      const float* srow = &scores[tid * SCORE_STRIDE];
      int nloc = min(ROWS_PER_ITER, chunkEnd - rowBase);
      for (int c = 0; c < nloc; ++c) {
        float v = srow[c];
        if (v > tv[TOPK - 1]) {
          int p = TOPK - 1;
          while (p > 0 && tv[p - 1] < v) {
            tv[p] = tv[p - 1]; ti[p] = ti[p - 1]; --p;
          }
          tv[p] = v; ti[p] = rowBase + c;
        }
      }
    }
    __syncthreads();
  }

  if (tid < QT) {
    int gq = qtile * QT + tid;
    float* ov = candVal + ((size_t)gq * nChunks + chunk) * TOPK;
    int*   oi = candIdx + ((size_t)gq * nChunks + chunk) * TOPK;
    for (int j = 0; j < TOPK; ++j) { ov[j] = tval[tid][j]; oi[j] = tidxs[tid][j]; }
  }
}

// ---------------------------------------------------------------------------
// Kernel 2: merge partial top-32 lists, softmax(log rbf + log h), gather-sum.
// One block (256 threads) per global query.
// ---------------------------------------------------------------------------
__global__ __launch_bounds__(256) void finalize_kernel(
    const float* __restrict__ candVal, const int* __restrict__ candIdx,
    const float* __restrict__ V, const float* __restrict__ h,
    const float* __restrict__ e,
    int nChunks, int DV, int DE,
    float* __restrict__ rV, float* __restrict__ rE,
    float* __restrict__ wOut, float* __restrict__ idxOut) {
  __shared__ float fv[TOPK];
  __shared__ int   fi[TOPK];
  __shared__ float wts[TOPK];
  const int gq  = blockIdx.x;
  const int tid = threadIdx.x;

  if (tid == 0) {
    for (int j = 0; j < TOPK; ++j) { fv[j] = -1e30f; fi[j] = 0; }
    for (int c = 0; c < nChunks; ++c) {
      const float* cv = candVal + ((size_t)gq * nChunks + c) * TOPK;
      const int*   ci = candIdx + ((size_t)gq * nChunks + c) * TOPK;
      for (int j = 0; j < TOPK; ++j) {
        float v = cv[j];
        if (v <= fv[TOPK - 1]) break;  // chunk lists sorted descending
        int p = TOPK - 1;
        while (p > 0 && fv[p - 1] < v) { fv[p] = fv[p - 1]; fi[p] = fi[p - 1]; --p; }
        fv[p] = v; fi[p] = ci[j];
      }
    }
    // rbf = exp(-(2-2s)/(2*0.5^2)) = exp(4(s-1)); softmax(log(rbf+eps)+log(h+eps))
    float mx = -1e30f;
    for (int k = 0; k < TOPK; ++k) {
      float w = __expf(4.0f * (fv[k] - 1.0f));
      float l = __logf(w + 1e-8f) + __logf(h[fi[k]] + 1e-8f);
      wts[k] = l;
      mx = fmaxf(mx, l);
    }
    float s = 0.f;
    for (int k = 0; k < TOPK; ++k) { float x = __expf(wts[k] - mx); wts[k] = x; s += x; }
    float inv = 1.0f / s;
    for (int k = 0; k < TOPK; ++k) wts[k] *= inv;
  }
  __syncthreads();

  for (int d = tid; d < DV; d += blockDim.x) {
    float a = 0.f;
#pragma unroll 8
    for (int k = 0; k < TOPK; ++k) a += wts[k] * V[(size_t)fi[k] * DV + d];
    rV[(size_t)gq * DV + d] = a;
  }
  for (int d = tid; d < DE; d += blockDim.x) {
    float a = 0.f;
    for (int k = 0; k < TOPK; ++k) a += wts[k] * e[(size_t)fi[k] * DE + d];
    rE[(size_t)gq * DE + d] = a;
  }
  if (tid < TOPK) {
    wOut[(size_t)gq * TOPK + tid]   = wts[tid];
    idxOut[(size_t)gq * TOPK + tid] = (float)fi[tid];
  }
}

// ---------------------------------------------------------------------------
// Host launcher
// ---------------------------------------------------------------------------
extern "C" void kernel_launch(void* const* d_in, const int* in_sizes, int n_in,
                              void* d_out, int out_size, void* d_ws, size_t ws_size,
                              hipStream_t stream) {
  const float* qf = (const float*)d_in[0];  // [B,T,128]
  const float* Kf = (const float*)d_in[1];  // [N,128]
  const float* Vf = (const float*)d_in[2];  // [N,DV]
  const float* hf = (const float*)d_in[3];  // [N]
  const float* ef = (const float*)d_in[4];  // [N,DE]
  const int N  = in_sizes[3];
  const int BT = in_sizes[0] / D;
  const int DV = in_sizes[2] / N;
  const int DE = in_sizes[4] / N;
  const int nChunks = NCHUNKS;
  const int chunkSize = (N + nChunks - 1) / nChunks;

  // Workspace layout
  char* ws = (char*)d_ws;
  size_t off = 0;
  bf16_t* qb = (bf16_t*)(ws + off);
  off += (size_t)BT * D * sizeof(bf16_t); off = (off + 255) & ~(size_t)255;
  bf16_t* Kb = (bf16_t*)(ws + off);
  off += (size_t)N * D * sizeof(bf16_t);  off = (off + 255) & ~(size_t)255;
  float* candVal = (float*)(ws + off);
  off += (size_t)BT * nChunks * TOPK * sizeof(float);
  int* candIdx = (int*)(ws + off);

  // Output layout: r_V | r_E | weights | topk_idx (flat, return order)
  float* rV     = (float*)d_out;
  float* rE     = rV + (size_t)BT * DV;
  float* wOut   = rE + (size_t)BT * DE;
  float* idxOut = wOut + (size_t)BT * TOPK;

  int nq = BT * D;
  cvt_bf16_kernel<<<(nq + 255) / 256, 256, 0, stream>>>(qf, qb, nq);
  int nk = N * D;
  cvt_bf16_kernel<<<(nk + 255) / 256, 256, 0, stream>>>(Kf, Kb, nk);

  dim3 g1(BT / QT, nChunks);
  sim_topk_kernel<<<g1, 128, 0, stream>>>(qb, Kb, N, chunkSize, nChunks,
                                          candVal, candIdx);

  finalize_kernel<<<BT, 256, 0, stream>>>(candVal, candIdx, Vf, hf, ef,
                                          nChunks, DV, DE, rV, rE, wOut, idxOut);
}
